// Syntax_HeteroSAGE_5815385719106
// MI455X (gfx1250) — matrix operand order repaired and checked by reading, not compiled, not run on hardware
//
#include <hip/hip_runtime.h>
#include <hip/hip_bf16.h>

#define DM 128
#define DD (DM * DM)

typedef __attribute__((ext_vector_type(16))) __bf16 v16bf;
typedef __attribute__((ext_vector_type(8)))  float  v8f;

union BF16x16 { v16bf v; unsigned u[8]; uint4 q[2]; };

__device__ __forceinline__ unsigned pack_bf16x2(float lo, float hi) {
  union { float f; unsigned u; } a, b;
  a.f = lo; b.f = hi;
  unsigned ra = a.u + 0x7FFFu + ((a.u >> 16) & 1u);   // round-to-nearest-even
  unsigned rb = b.u + 0x7FFFu + ((b.u >> 16) & 1u);
  return (ra >> 16) | (rb & 0xFFFF0000u);
}

// monotone float <-> uint mapping for atomicMax-based segment max
__device__ __forceinline__ unsigned encF(float f) {
  union { float f; unsigned u; } v; v.f = f;
  return (v.u & 0x80000000u) ? ~v.u : (v.u | 0x80000000u);
}
__device__ __forceinline__ float decF(unsigned u) {
  union { float f; unsigned u; } v;
  v.u = (u & 0x80000000u) ? (u ^ 0x80000000u) : ~u;
  return v.f;
}
#define ENC_NEG_INF 0x007FFFFFu   // encF(-inf)

// ---------------------------------------------------------------------------
// GEMM: C[n,128] = preOp(A[n,128]) @ B[128,128] (+bias), bf16 WMMA, f32 accum
// preOp(x) = preRelu ? max(x*preScale, 0) : x
// Block: 256 threads (8 waves); block tile = 128 rows; wave tile = 16 x 128.
// B is staged into LDS in *fragment-major* order: fragment (ks,ct) for lane l
// occupies 8 contiguous dwords -> one fragment = 2 x ds_load_b128.
// ---------------------------------------------------------------------------
__global__ void __launch_bounds__(256)
gemm_bf16_wmma(const float* __restrict__ A, const float* __restrict__ B,
               float* __restrict__ C, const float* __restrict__ bias,
               int nRows, float preScale, int preRelu)
{
  // [frag(=ks*8+ct)][lane][v] : 32*32*8 dwords = 32KB of packed bf16 pairs
  __shared__ unsigned Blds[32 * 32 * 8];

  const int tid = threadIdx.x;
  for (int idx = tid; idx < 8192; idx += 256) {
    const int v    = idx & 7;
    const int lane = (idx >> 3) & 31;
    const int f    = idx >> 8;           // 0..31
    const int ks   = f >> 3;
    const int ct   = f & 7;
    const int n    = ct * 16 + (lane & 15);
    const int k    = ks * 32 + ((lane >> 4) << 3) + 2 * v + ((v >> 2) << 3);  // B frag: K pairs
    Blds[idx] = pack_bf16x2(B[k * DM + n], B[(k + 1) * DM + n]);
  }
  __syncthreads();

  const int wave = tid >> 5;
  const int lane = tid & 31;
  const int half = lane >> 4;
  const int l16  = lane & 15;
  const int rowBase = blockIdx.x * 128 + wave * 16;

  int aRow = rowBase + l16;
  if (aRow >= nRows) aRow = nRows - 1;      // clamp; OOB results masked on store
  const float* aPtr = A + (size_t)aRow * DM;

  const uint4* bq = (const uint4*)Blds;

  v8f acc[8] = {};

#pragma unroll
  for (int ks = 0; ks < 4; ++ks) {
    const int kb = ks * 32;
    // A fragment (16x32 bf16): per lane two contiguous 8-float runs
    float4 c0 = ((const float4*)(aPtr + kb + 8 * half))[0];
    float4 c1 = ((const float4*)(aPtr + kb + 8 * half))[1];
    float4 c2 = ((const float4*)(aPtr + kb + 16 + 8 * half))[0];
    float4 c3 = ((const float4*)(aPtr + kb + 16 + 8 * half))[1];
    if (preRelu) {
      c0.x = fmaxf(c0.x * preScale, 0.f); c0.y = fmaxf(c0.y * preScale, 0.f);
      c0.z = fmaxf(c0.z * preScale, 0.f); c0.w = fmaxf(c0.w * preScale, 0.f);
      c1.x = fmaxf(c1.x * preScale, 0.f); c1.y = fmaxf(c1.y * preScale, 0.f);
      c1.z = fmaxf(c1.z * preScale, 0.f); c1.w = fmaxf(c1.w * preScale, 0.f);
      c2.x = fmaxf(c2.x * preScale, 0.f); c2.y = fmaxf(c2.y * preScale, 0.f);
      c2.z = fmaxf(c2.z * preScale, 0.f); c2.w = fmaxf(c2.w * preScale, 0.f);
      c3.x = fmaxf(c3.x * preScale, 0.f); c3.y = fmaxf(c3.y * preScale, 0.f);
      c3.z = fmaxf(c3.z * preScale, 0.f); c3.w = fmaxf(c3.w * preScale, 0.f);
    }
    BF16x16 afrag;
    afrag.u[0] = pack_bf16x2(c0.x, c0.y);
    afrag.u[1] = pack_bf16x2(c0.z, c0.w);
    afrag.u[2] = pack_bf16x2(c1.x, c1.y);
    afrag.u[3] = pack_bf16x2(c1.z, c1.w);
    afrag.u[4] = pack_bf16x2(c2.x, c2.y);
    afrag.u[5] = pack_bf16x2(c2.z, c2.w);
    afrag.u[6] = pack_bf16x2(c3.x, c3.y);
    afrag.u[7] = pack_bf16x2(c3.z, c3.w);

#pragma unroll
    for (int ct = 0; ct < 8; ++ct) {
      BF16x16 bfrag;
      const int fi = ((ks * 8 + ct) * 32 + lane) * 2;   // uint4 index
      bfrag.q[0] = bq[fi];
      bfrag.q[1] = bq[fi + 1];
      acc[ct] = __builtin_amdgcn_wmma_f32_16x16x32_bf16(
          false, afrag.v, false, bfrag.v, (short)0, acc[ct], false, false);
    }
  }

#pragma unroll
  for (int ct = 0; ct < 8; ++ct) {
    const int n  = ct * 16 + l16;
    const float bv = bias ? bias[n] : 0.f;
#pragma unroll
    for (int v = 0; v < 8; ++v) {
      const int row = rowBase + v + 8 * half;
      if (row < nRows) C[(size_t)row * DM + n] = acc[ct][v] + bv;
    }
  }
}

// ---------------------------------------------------------------------------
// out[i] = dot(H[i,:], vec[:])  (one wave per row)
// ---------------------------------------------------------------------------
__global__ void __launch_bounds__(256)
rowdot_kernel(const float* __restrict__ H, const float* __restrict__ vec,
              float* __restrict__ out, int n)
{
  const int wid  = (blockIdx.x * 256 + threadIdx.x) >> 5;
  const int lane = threadIdx.x & 31;
  if (wid >= n) return;
  const float4 h = ((const float4*)(H + (size_t)wid * DM))[lane];
  const float4 a = ((const float4*)vec)[lane];
  float s = h.x * a.x + h.y * a.y + h.z * a.z + h.w * a.w;
#pragma unroll
  for (int off = 16; off; off >>= 1) s += __shfl_xor(s, off, 32);
  if (lane == 0) out[wid] = s;
}

// ---------------------------------------------------------------------------
// Edge passes
// ---------------------------------------------------------------------------
__global__ void __launch_bounds__(256)
edge_logits(const float* __restrict__ es, const float* __restrict__ ed,
            const int* __restrict__ sidx, const int* __restrict__ didx,
            float* __restrict__ ebuf, unsigned* __restrict__ menc, int E)
{
  const int k = blockIdx.x * 256 + threadIdx.x;
  if (k >= E) return;
  float e = es[sidx[k]] + ed[didx[k]];
  e = e > 0.f ? e : 0.2f * e;               // leaky_relu(0.2)
  ebuf[k] = e;
  atomicMax(&menc[didx[k]], encF(e));
}

__global__ void __launch_bounds__(256)
edge_exp(float* __restrict__ ebuf, const unsigned* __restrict__ menc,
         float* __restrict__ denom, const int* __restrict__ didx, int E)
{
  const int k = blockIdx.x * 256 + threadIdx.x;
  if (k >= E) return;
  const int d = didx[k];
  const float ex = __expf(ebuf[k] - decF(menc[d]));
  ebuf[k] = ex;
  atomicAdd(&denom[d], ex);
}

// one wave per edge: out[dst,:] += alpha * h_src[src,:]
__global__ void __launch_bounds__(256)
edge_aggregate(const float* __restrict__ ebuf, const float* __restrict__ denom,
               const float* __restrict__ hsrc, const int* __restrict__ sidx,
               const int* __restrict__ didx, float* __restrict__ out, int E)
{
  const int w    = (blockIdx.x * 256 + threadIdx.x) >> 5;
  const int lane = threadIdx.x & 31;
  if (w >= E) return;
  const int s = sidx[w], d = didx[w];
  const float alpha = ebuf[w] / denom[d];
  const float4 h = ((const float4*)(hsrc + (size_t)s * DM))[lane];
  float* o = out + (size_t)d * DM + lane * 4;
  atomicAdd(o + 0, alpha * h.x);
  atomicAdd(o + 1, alpha * h.y);
  atomicAdd(o + 2, alpha * h.z);
  atomicAdd(o + 3, alpha * h.w);
}

// ---------------------------------------------------------------------------
// Init helpers
// ---------------------------------------------------------------------------
__global__ void fill_u32(unsigned* p, unsigned v, int n)
{ int i = blockIdx.x * 256 + threadIdx.x; if (i < n) p[i] = v; }

__global__ void fill_f32(float* p, float v, int n)
{ int i = blockIdx.x * 256 + threadIdx.x; if (i < n) p[i] = v; }

__global__ void init_rows_bias(float* __restrict__ out, const float* b0,
                               const float* b1, const float* b2, int nElems)
{
  const int i = blockIdx.x * 256 + threadIdx.x;
  if (i >= nElems) return;
  const int c = i & 127;
  float v = b0[c];
  if (b1) v += b1[c];
  if (b2) v += b2[c];
  out[i] = v;
}

// ---------------------------------------------------------------------------
extern "C" void kernel_launch(void* const* d_in, const int* in_sizes, int n_in,
                              void* d_out, int out_size, void* d_ws, size_t ws_size,
                              hipStream_t stream)
{
  const float* x_word   = (const float*)d_in[0];
  const float* x_feat[3]= {(const float*)d_in[1], (const float*)d_in[2], (const float*)d_in[3]};
  const int*   srcs[3]  = {(const int*)d_in[4], (const int*)d_in[6], (const int*)d_in[8]};
  const int*   dsts[3]  = {(const int*)d_in[5], (const int*)d_in[7], (const int*)d_in[9]};
  const float* W1  = (const float*)d_in[10];
  const float* as1 = (const float*)d_in[11];
  const float* ad1 = (const float*)d_in[12];
  const float* b1  = (const float*)d_in[13];
  const float* W2  = (const float*)d_in[14];
  const float* as2 = (const float*)d_in[15];
  const float* ad2 = (const float*)d_in[16];
  const float* b2  = (const float*)d_in[17];
  const float* Wfc = (const float*)d_in[18];
  const float* bfc = (const float*)d_in[19];
  float* out = (float*)d_out;                 // also used as layer-2 word accumulator

  const int NW = in_sizes[0] / DM;            // 100000
  const int NF = in_sizes[1] / DM;            // 1000
  const int E  = in_sizes[4];                 // 400000

  // ---- workspace carve ----
  char* wsp = (char*)d_ws;
  auto carve = [&](size_t bytes) { void* p = wsp; wsp += (bytes + 255) & ~(size_t)255; return p; };
  float*    hbuf  = (float*)carve((size_t)NW * DM * 4);  // word projection (reused)
  float*    wA    = (float*)carve((size_t)NW * DM * 4);  // layer-1 word accumulator
  float*    fo[3];
  for (int r = 0; r < 3; ++r) fo[r] = (float*)carve((size_t)NF * DM * 4);
  float*    hf    = (float*)carve((size_t)NF * DM * 4);  // feat projection (reused)
  float*    esw   = (float*)carve((size_t)NW * 4);
  float*    edw   = (float*)carve((size_t)NW * 4);
  float*    ef1   = (float*)carve((size_t)NF * 4);
  float*    ef2   = (float*)carve((size_t)NF * 4);
  float*    ebuf  = (float*)carve((size_t)E * 4);
  unsigned* mencW = (unsigned*)carve((size_t)NW * 4);
  float*    denW  = (float*)carve((size_t)NW * 4);
  unsigned* mencF = (unsigned*)carve((size_t)NF * 4);
  float*    denF  = (float*)carve((size_t)NF * 4);

  auto gemm = [&](float* C, const float* A, const float* B, const float* bias,
                  int n, float scale, int relu) {
    gemm_bf16_wmma<<<dim3((n + 127) / 128), dim3(256), 0, stream>>>(A, B, C, bias, n, scale, relu);
  };
  auto rowdot = [&](const float* H, const float* v, float* o, int n) {
    rowdot_kernel<<<dim3((n + 7) / 8), dim3(256), 0, stream>>>(H, v, o, n);
  };
  auto eA = [&](const float* es, const float* ed, const int* s, const int* d, unsigned* m) {
    edge_logits<<<dim3((E + 255) / 256), dim3(256), 0, stream>>>(es, ed, s, d, ebuf, m, E);
  };
  auto eB = [&](const unsigned* m, float* den, const int* d) {
    edge_exp<<<dim3((E + 255) / 256), dim3(256), 0, stream>>>(ebuf, m, den, d, E);
  };
  auto eC = [&](const float* den, const float* hs, const int* s, const int* d, float* o) {
    edge_aggregate<<<dim3((E + 7) / 8), dim3(256), 0, stream>>>(ebuf, den, hs, s, d, o, E);
  };
  auto fillU = [&](unsigned* p, unsigned v, int n) {
    fill_u32<<<dim3((n + 255) / 256), dim3(256), 0, stream>>>(p, v, n);
  };
  auto fillF = [&](float* p, float v, int n) {
    fill_f32<<<dim3((n + 255) / 256), dim3(256), 0, stream>>>(p, v, n);
  };
  auto initB = [&](float* o, const float* bb0, const float* bb1, const float* bb2, int n) {
    init_rows_bias<<<dim3((n * DM + 255) / 256), dim3(256), 0, stream>>>(o, bb0, bb1, bb2, n * DM);
  };

  // ======================= Layer 1 =======================
  for (int r = 0; r < 3; ++r) {
    const float* Wf = W1 + (size_t)r * DD;        // word -> feat_r
    const float* Wb = W1 + (size_t)(3 + r) * DD;  // feat_r -> word

    // forward GAT: src = word, dst = feat_r
    gemm(hbuf, x_word, Wf, nullptr, NW, 1.f, 0);
    rowdot(hbuf, as1 + r * DM, esw, NW);
    gemm(hf, x_feat[r], Wf, nullptr, NF, 1.f, 0);
    rowdot(hf, ad1 + r * DM, ef2, NF);
    fillU(mencF, ENC_NEG_INF, NF);
    fillF(denF, 0.f, NF);
    initB(fo[r], b1 + r * DM, nullptr, nullptr, NF);
    eA(esw, ef2, srcs[r], dsts[r], mencF);
    eB(mencF, denF, dsts[r]);
    eC(denF, hbuf, srcs[r], dsts[r], fo[r]);

    // reverse GAT: src = feat_r, dst = word
    gemm(hf, x_feat[r], Wb, nullptr, NF, 1.f, 0);
    rowdot(hf, as1 + (3 + r) * DM, ef1, NF);
    gemm(hbuf, x_word, Wb, nullptr, NW, 1.f, 0);
    rowdot(hbuf, ad1 + (3 + r) * DM, edw, NW);
    if (r == 0) initB(wA, b1 + 3 * DM, b1 + 4 * DM, b1 + 5 * DM, NW);
    fillU(mencW, ENC_NEG_INF, NW);
    fillF(denW, 0.f, NW);
    eA(ef1, edw, dsts[r], srcs[r], mencW);     // edge src index = feat, dst index = word
    eB(mencW, denW, srcs[r]);
    eC(denW, hf, dsts[r], srcs[r], wA);
  }
  // word layer-1 output = relu(wA / 3): fused into layer-2 GEMM loads.
  // feat layer-1 outputs = relu(fo[r]):  fused into layer-2 GEMM loads.

  // ======================= Layer 2 (only feat -> word GATs feed the output) ==
  for (int r = 0; r < 3; ++r) {
    const float* Wb = W2 + (size_t)(3 + r) * DD;

    gemm(hf, fo[r], Wb, nullptr, NF, 1.f, 1);            // relu(fo) @ W
    rowdot(hf, as2 + (3 + r) * DM, ef1, NF);
    gemm(hbuf, wA, Wb, nullptr, NW, 1.f / 3.f, 1);       // relu(wA/3) @ W
    rowdot(hbuf, ad2 + (3 + r) * DM, edw, NW);
    if (r == 0) initB(out, b2 + 3 * DM, b2 + 4 * DM, b2 + 5 * DM, NW);
    fillU(mencW, ENC_NEG_INF, NW);
    fillF(denW, 0.f, NW);
    eA(ef1, edw, dsts[r], srcs[r], mencW);
    eB(mencW, denW, srcs[r]);
    eC(denW, hf, dsts[r], srcs[r], out);                 // accumulate into d_out
  }

  // ======================= Final FC (in place: row-block i reads only rows i) =
  gemm(out, out, Wfc, bfc, NW, 1.f / 3.f, 1);            // relu(out/3) @ Wfc + bfc
}
